// Self_attention_pixel_wise_39367670235286
// MI455X (gfx1250) — compile-verified, matrix operand order checked by reading, not run on hardware
//
#include <hip/hip_runtime.h>

#define BATCH 4
#define NPIX  4096
#define CIN   256
#define CPRJ  128
#define L2E   1.44269504088896340736f

typedef __attribute__((ext_vector_type(16))) _Float16 v16h;
typedef __attribute__((ext_vector_type(8)))  float    v8f;
typedef __attribute__((ext_vector_type(4)))  float    v4f;
typedef __attribute__((ext_vector_type(4)))  unsigned int v4u;

union FragU { v16h h; v4u q[2]; };

// Load a 16-half WMMA fragment as two contiguous 16B chunks.
static __device__ inline v16h ld_frag(const _Float16* p0, const _Float16* p1) {
    FragU f;
    f.q[0] = *reinterpret_cast<const v4u*>(p0);
    f.q[1] = *reinterpret_cast<const v4u*>(p1);
    return f.h;
}

// ---------------------------------------------------------------------------
// Kernel 0: pack Wk/Wq/Wv/Ws (f32 row-major [256][128]) into B-fragment-ready
// f16 layout: Wf[(t*64 + kstep*8 + j)*32 + lane][16 halves].
// B 32x16 layout: lane holds col n = j*16 + lane%16; half h holds
// k_local = (lane<16?0:8) + (h<8 ? h : h+8).
// ---------------------------------------------------------------------------
__global__ void swizzle_weights_kernel(const float* __restrict__ Wk,
                                       const float* __restrict__ Wq,
                                       const float* __restrict__ Wv,
                                       const float* __restrict__ Ws,
                                       _Float16* __restrict__ Wf) {
    int idx  = blockIdx.x * blockDim.x + threadIdx.x;   // 0..8191
    int lane = idx & 31;
    int frag = idx >> 5;                                // t*64 + ks*8 + j
    int j    = frag & 7;
    int ks   = (frag >> 3) & 7;
    int t    = frag >> 6;
    const float* W = (t == 0) ? Wk : (t == 1) ? Wq : (t == 2) ? Wv : Ws;
    int n   = j * 16 + (lane & 15);
    int kb  = (lane < 16) ? 0 : 8;
    _Float16* dst = Wf + ((size_t)frag * 32 + lane) * 16;
#pragma unroll
    for (int h = 0; h < 16; ++h) {
        int kl = kb + ((h < 8) ? h : (h + 8));
        dst[h] = (_Float16)W[(size_t)(ks * 32 + kl) * CPRJ + n];
    }
}

// ---------------------------------------------------------------------------
// Kernel 1: projections. Grid = B*N/16 = 1024 blocks of 128 threads (4 waves).
// Wave t in {K,Q,V,S} computes its 16x128 output strip with 8x8 = 64 WMMAs.
// K,Q -> f16 row-major [b][n][128]; V -> f16 transposed [b][c'][n]; S -> f32.
// ---------------------------------------------------------------------------
__global__ void proj_kernel(const float* __restrict__ x,
                            const float* __restrict__ bk, const float* __restrict__ bq,
                            const float* __restrict__ bv, const float* __restrict__ bs,
                            const _Float16* __restrict__ Wf,
                            _Float16* __restrict__ Kb, _Float16* __restrict__ Qb,
                            _Float16* __restrict__ VT, float* __restrict__ Sb) {
    int tile = blockIdx.x;                  // 0..1023
    int b    = tile >> 8;
    int i    = tile & 255;
    // readfirstlane: wave id is uniform -> scalar branches, EXEC stays full.
    int wave = __builtin_amdgcn_readfirstlane(threadIdx.x >> 5);
    int lane = threadIdx.x & 31;
    int nn   = lane & 15;
    int kb8  = (lane < 16) ? 0 : 8;
    int hi8  = (lane < 16) ? 0 : 8;
    int row  = i * 16 + nn;

    // A fragments of x: 8 k-steps of 32, f32 -> f16 on the fly.
    const float* xrow = x + ((size_t)b * NPIX + row) * CIN;
    v16h a[8];
#pragma unroll
    for (int ks = 0; ks < 8; ++ks) {
        const float* p = xrow + ks * 32 + kb8;
        v4f x0 = *reinterpret_cast<const v4f*>(p);
        v4f x1 = *reinterpret_cast<const v4f*>(p + 4);
        v4f x2 = *reinterpret_cast<const v4f*>(p + 16);
        v4f x3 = *reinterpret_cast<const v4f*>(p + 20);
        FragU f;
        _Float16* hp = reinterpret_cast<_Float16*>(&f);
#pragma unroll
        for (int e = 0; e < 4; ++e) {
            hp[e]      = (_Float16)x0[e];
            hp[4 + e]  = (_Float16)x1[e];
            hp[8 + e]  = (_Float16)x2[e];
            hp[12 + e] = (_Float16)x3[e];
        }
        a[ks] = f.h;
    }

    const float* bias = (wave == 0) ? bk : (wave == 1) ? bq : (wave == 2) ? bv : bs;

#pragma unroll 1
    for (int j = 0; j < 8; ++j) {
        // Preload all 8 weight fragments -> one big load clause, then let the
        // WMMA chain drain the loads with progressive loadcnt waits.
        const _Float16* wbase = Wf + (((size_t)wave * 64 + j) * 32 + lane) * 16;
        v16h bf[8];
#pragma unroll
        for (int ks = 0; ks < 8; ++ks) {
            const _Float16* wp = wbase + (size_t)ks * 8 * 32 * 16;
            bf[ks] = ld_frag(wp, wp + 8);
        }
        v8f c = {};
#pragma unroll
        for (int ks = 0; ks < 8; ++ks)
            c = __builtin_amdgcn_wmma_f32_16x16x32_f16(false, a[ks], false, bf[ks],
                                                       (short)0, c, false, false);
        int n = j * 16 + nn;
        float bval = bias[n];
#pragma unroll
        for (int r = 0; r < 8; ++r) {
            int   m    = r + hi8;
            int   grow = i * 16 + m;
            float val  = c[r] + bval;
            if (wave == 0)      Kb[((size_t)b * NPIX + grow) * CPRJ + n] = (_Float16)val;
            else if (wave == 1) Qb[((size_t)b * NPIX + grow) * CPRJ + n] = (_Float16)val;
            else if (wave == 2) VT[((size_t)b * CPRJ + n) * NPIX + grow] = (_Float16)val;
            else                Sb[((size_t)b * NPIX + grow) * CPRJ + n] = val;
        }
    }
}

// ---------------------------------------------------------------------------
// Kernel 2: flash attention. Grid = 256 blocks x 128 threads; each wave owns a
// 16-row K tile and streams 64-column Q/V blocks (wide tile amortizes the
// softmax VALU work against the co-executing WMMA pipe). Scores stay in regs.
// ---------------------------------------------------------------------------
__global__ void attn_kernel(const _Float16* __restrict__ Kb,
                            const _Float16* __restrict__ Qb,
                            const _Float16* __restrict__ VT,
                            const float* __restrict__ Sb,
                            float* __restrict__ out) {
    __shared__ __attribute__((aligned(64))) _Float16 Pbuf[4][16 * 64];

    int wave = __builtin_amdgcn_readfirstlane(threadIdx.x >> 5);
    int lane = threadIdx.x & 31;
    int tile = blockIdx.x * 4 + wave;       // 0..1023
    int b    = tile >> 8;
    int i    = tile & 255;
    int nn   = lane & 15;
    int kb8  = (lane < 16) ? 0 : 8;
    int hi8  = (lane < 16) ? 0 : 8;

    const _Float16* Kbase = Kb + (size_t)b * NPIX * CPRJ;
    const _Float16* Qbase = Qb + (size_t)b * NPIX * CPRJ;
    const _Float16* Vbase = VT + (size_t)b * CPRJ * NPIX;

    // K A-fragments (row = i*16 + lane%16), 4 k-steps of 32.
    v16h aK[4];
    {
        const _Float16* krow = Kbase + (size_t)(i * 16 + nn) * CPRJ;
#pragma unroll
        for (int ks = 0; ks < 4; ++ks) {
            const _Float16* p = krow + ks * 32 + kb8;
            aK[ks] = ld_frag(p, p + 16);
        }
    }

    v8f zero = {};
    v8f o[8];
#pragma unroll
    for (int j = 0; j < 8; ++j) o[j] = zero;
    float mrun[8], lrun[8];
#pragma unroll
    for (int r = 0; r < 8; ++r) { mrun[r] = -1e30f; lrun[r] = 0.0f; }

    _Float16* Pw = &Pbuf[wave][0];

#pragma unroll 1
    for (int J = 0; J < NPIX / 64; ++J) {
        // ---- scores: four 16x16 tiles over the 128-deep K dim --------------
        // k-step outer / tile inner: each group of 4 WMMAs is independent
        // (distinct accumulators), so the next k-step's loads overlap them.
        v8f c[4];
#pragma unroll
        for (int t = 0; t < 4; ++t) c[t] = zero;
        const _Float16* qb = Qbase + (size_t)(J * 64 + nn) * CPRJ + kb8;
#pragma unroll
        for (int ks = 0; ks < 4; ++ks) {
            v16h bq[4];
#pragma unroll
            for (int t = 0; t < 4; ++t) {
                const _Float16* p = qb + (size_t)t * 16 * CPRJ + ks * 32;
                bq[t] = ld_frag(p, p + 16);
            }
#pragma unroll
            for (int t = 0; t < 4; ++t)
                c[t] = __builtin_amdgcn_wmma_f32_16x16x32_f16(false, aK[ks], false, bq[t],
                                                              (short)0, c[t], false, false);
        }

        // ---- online softmax (row stats via wave32 shfl_xor in 16-lane halves)
#pragma unroll
        for (int r = 0; r < 8; ++r) {
            float mr = fmaxf(fmaxf(c[0][r], c[1][r]), fmaxf(c[2][r], c[3][r]));
#pragma unroll
            for (int msk = 8; msk >= 1; msk >>= 1)
                mr = fmaxf(mr, __shfl_xor(mr, msk, 32));
            float mnew  = fmaxf(mrun[r], mr);
            float scale = exp2f((mrun[r] - mnew) * L2E);
            float p0 = exp2f((c[0][r] - mnew) * L2E);
            float p1 = exp2f((c[1][r] - mnew) * L2E);
            float p2 = exp2f((c[2][r] - mnew) * L2E);
            float p3 = exp2f((c[3][r] - mnew) * L2E);
            float rs = (p0 + p1) + (p2 + p3);
#pragma unroll
            for (int msk = 8; msk >= 1; msk >>= 1)
                rs += __shfl_xor(rs, msk, 32);
            lrun[r] = lrun[r] * scale + rs;
            mrun[r] = mnew;
#pragma unroll
            for (int j = 0; j < 8; ++j) o[j][r] *= scale;
            int m = r + hi8;                        // C layout -> row index
            Pw[m * 64 + nn]      = (_Float16)p0;    // P row-major [16][64]
            Pw[m * 64 + 16 + nn] = (_Float16)p1;
            Pw[m * 64 + 32 + nn] = (_Float16)p2;
            Pw[m * 64 + 48 + nn] = (_Float16)p3;
        }

        // DS pipeline is per-wave in-order; wait + fence before re-reading P.
        asm volatile("s_wait_dscnt 0x0" ::: "memory");

        // ---- P (16x64 f16 = two A-fragments from LDS) times V block --------
        v16h aP0, aP1;
        {
            const _Float16* pp = Pw + nn * 64 + kb8;
            aP0 = ld_frag(pp, pp + 16);
            aP1 = ld_frag(pp + 32, pp + 48);
        }
#pragma unroll
        for (int j = 0; j < 8; ++j) {
            const _Float16* vp = Vbase + (size_t)(j * 16 + nn) * NPIX + J * 64 + kb8;
            v16h bV0 = ld_frag(vp, vp + 16);
            o[j] = __builtin_amdgcn_wmma_f32_16x16x32_f16(false, aP0, false, bV0,
                                                          (short)0, o[j], false, false);
            v16h bV1 = ld_frag(vp + 32, vp + 48);
            o[j] = __builtin_amdgcn_wmma_f32_16x16x32_f16(false, aP1, false, bV1,
                                                          (short)0, o[j], false, false);
        }
        asm volatile("" ::: "memory");
    }

    // ---- epilogue: normalize, add shortcut, store f32 ----------------------
    float inv[8];
#pragma unroll
    for (int r = 0; r < 8; ++r) inv[r] = 1.0f / lrun[r];
#pragma unroll
    for (int j = 0; j < 8; ++j) {
#pragma unroll
        for (int r = 0; r < 8; ++r) {
            int    m   = r + hi8;
            size_t idx = ((size_t)b * NPIX + i * 16 + m) * CPRJ + j * 16 + nn;
            out[idx]   = o[j][r] * inv[r] + Sb[idx];
        }
    }
}

// ---------------------------------------------------------------------------
extern "C" void kernel_launch(void* const* d_in, const int* in_sizes, int n_in,
                              void* d_out, int out_size, void* d_ws, size_t ws_size,
                              hipStream_t stream) {
    const float* x  = (const float*)d_in[0];
    const float* Wk = (const float*)d_in[1];
    const float* bk = (const float*)d_in[2];
    const float* Wq = (const float*)d_in[3];
    const float* bq = (const float*)d_in[4];
    const float* Wv = (const float*)d_in[5];
    const float* bv = (const float*)d_in[6];
    const float* Ws = (const float*)d_in[7];
    const float* bs = (const float*)d_in[8];
    float* out = (float*)d_out;

    char* ws = (char*)d_ws;
    const size_t szWf = 256u * 32u * 16u * sizeof(_Float16);            // 256 KB
    const size_t szKQ = (size_t)BATCH * NPIX * CPRJ * sizeof(_Float16); // 4 MB each
    _Float16* Wf = (_Float16*)ws;
    _Float16* Kb = (_Float16*)(ws + szWf);
    _Float16* Qb = (_Float16*)(ws + szWf + szKQ);
    _Float16* VT = (_Float16*)(ws + szWf + 2 * szKQ);
    float*    Sb = (float*)   (ws + szWf + 3 * szKQ);

    swizzle_weights_kernel<<<32, 256, 0, stream>>>(Wk, Wq, Wv, Ws, Wf);
    proj_kernel<<<BATCH * (NPIX / 16), 128, 0, stream>>>(x, bk, bq, bv, bs, Wf,
                                                         Kb, Qb, VT, Sb);
    attn_kernel<<<BATCH * (NPIX / 16) / 4, 128, 0, stream>>>(Kb, Qb, VT, Sb, out);
}
